// MultiHeadAttention_76020921139935
// MI455X (gfx1250) — matrix operand order repaired and checked by reading, not compiled, not run on hardware
//
#include <hip/hip_runtime.h>
#include <math.h>
#include <stdint.h>

#define D   1024
#define H   16
#define DH  64
#define NCTX 2048
#define NB  2

typedef __attribute__((ext_vector_type(16))) __bf16 v16bf;
typedef __attribute__((ext_vector_type(8)))  __bf16 v8bf;
typedef __attribute__((ext_vector_type(8)))  float  v8f;
typedef int v4i_ __attribute__((vector_size(16)));

__device__ __forceinline__ v8f wmma_bf16(v16bf a, v16bf b, v8f c) {
    return __builtin_amdgcn_wmma_f32_16x16x32_bf16(false, a, false, b,
                                                   (short)0, c, false, false);
}

// ---- CDNA5 async global->LDS copy (ASYNCcnt path), with safe fallback -------
#if __has_builtin(__builtin_amdgcn_global_load_async_to_lds_b128)
typedef __attribute__((address_space(1))) v4i_* as1_v4i;
typedef __attribute__((address_space(3))) v4i_* as3_v4i;
#endif

__device__ __forceinline__ void async_copy_b128(const __bf16* g, __bf16* l) {
#if __has_builtin(__builtin_amdgcn_global_load_async_to_lds_b128)
    __builtin_amdgcn_global_load_async_to_lds_b128(
        (as1_v4i)(uintptr_t)g,
        (as3_v4i)(unsigned int)(uintptr_t)l, 0, 0);
#else
    *(v8bf*)l = *(const v8bf*)g;
#endif
}
__device__ __forceinline__ void async_wait0() {
#if __has_builtin(__builtin_amdgcn_global_load_async_to_lds_b128)
#if __has_builtin(__builtin_amdgcn_s_wait_asynccnt)
    __builtin_amdgcn_s_wait_asynccnt(0);
#else
    asm volatile("s_wait_asynccnt 0x0" ::: "memory");
#endif
#endif
}

// A-fragment (16x32, M x K): lane holds row M=lane%16, K = kb..kb+7 and kb+16..kb+23
// with kb = (lane/16)*8.  base points at element [0][0] of the 16x32 tile.
__device__ __forceinline__ v16bf frag_MK(const __bf16* base, int stride, int lane) {
    const __bf16* p = base + (size_t)(lane & 15) * stride + ((lane >> 4) * 8);
    union { v16bf v; v8bf h[2]; } u;
    u.h[0] = *(const v8bf*)p;
    u.h[1] = *(const v8bf*)(p + 16);
    return u.v;
}

// B-fragment from Bt (N x K row-major): lane holds col N=lane%16,
// K = kb..kb+7 and kb+16..kb+23 with kb = (lane/16)*8.
__device__ __forceinline__ v16bf frag_NK(const __bf16* base, int stride, int lane) {
    const __bf16* p = base + (size_t)(lane & 15) * stride + ((lane >> 4) * 8);
    union { v16bf v; v8bf h[2]; } u;
    u.h[0] = *(const v8bf*)p;
    u.h[1] = *(const v8bf*)(p + 16);
    return u.v;
}

// ---------------- one-shot f32 -> bf16 conversion (x and weights) ------------
__global__ void __launch_bounds__(256)
cvt_kernel(const float* __restrict__ src, __bf16* __restrict__ dst, int n) {
    int i = (blockIdx.x * 256 + threadIdx.x) * 8;
    if (i >= n) return;
    float4 a = ((const float4*)(src + i))[0];
    float4 b = ((const float4*)(src + i))[1];
    v8bf o;
    o[0] = (__bf16)a.x; o[1] = (__bf16)a.y; o[2] = (__bf16)a.z; o[3] = (__bf16)a.w;
    o[4] = (__bf16)b.x; o[5] = (__bf16)b.y; o[6] = (__bf16)b.z; o[7] = (__bf16)b.w;
    *(v8bf*)(dst + i) = o;
}

// ---------------- QKV projection: out = x @ W.T + b (all-bf16 GEMM) ----------
// mode 0: out[((b*H+h)*NCTX + n)*DH + dh]   (Q, K row-major per head)
// mode 1: out[((b*H+h)*DH + dh)*NCTX + n]   (V transposed per head)
__global__ void __launch_bounds__(128)
proj_kernel(const __bf16* __restrict__ x, const __bf16* __restrict__ W,
            const float* __restrict__ bias, __bf16* __restrict__ out, int mode) {
    __shared__ __bf16 lB[64][56];                 // 64 N-rows x 32 K (+24 pad)
    const int tid = threadIdx.x, lane = tid & 31, wv = tid >> 5;
    const int m0 = blockIdx.x * 64, n0 = blockIdx.y * 64;
    v8f acc[4] = {};
    const int lr = tid >> 1, lc = (tid & 1) * 16;     // cooperative W loader
    const __bf16* abase = x + (size_t)(m0 + wv * 16) * D;
    const __bf16* wrow  = W + (size_t)(n0 + lr) * D + lc;
    __bf16* ldst = &lB[lr][lc];

    for (int k0 = 0; k0 < D; k0 += 32) {
        __syncthreads();
        async_copy_b128(wrow + k0,     ldst);          // 16 bf16 = 2 x b128
        async_copy_b128(wrow + k0 + 8, ldst + 8);
        async_wait0();
        __syncthreads();
        v16bf af = frag_MK(abase + k0, D, lane);
        for (int jn = 0; jn < 4; ++jn)
            acc[jn] = wmma_bf16(af, frag_NK(&lB[jn * 16][0], 56, lane), acc[jn]);
    }

    const int col = lane & 15, halfOff = (lane < 16) ? 0 : 8;
    for (int jn = 0; jn < 4; ++jn)
        for (int r = 0; r < 8; ++r) {
            int gm = m0 + wv * 16 + r + halfOff;
            int gn = n0 + jn * 16 + col;
            float v = acc[jn][r] + bias[gn];
            int b = gm >> 11, nt = gm & (NCTX - 1), hh = gn >> 6, dh = gn & (DH - 1);
            size_t idx = (mode == 0)
                ? ((size_t)((b * H + hh) * NCTX + nt) * DH + dh)
                : ((size_t)((b * H + hh) * DH + dh) * NCTX + nt);
            out[idx] = (__bf16)v;
        }
}

// ---------------- flash attention over one 64-query block ---------------------
__global__ void __launch_bounds__(128)
attn_kernel(const __bf16* __restrict__ q, const __bf16* __restrict__ k,
            const __bf16* __restrict__ vt, __bf16* __restrict__ ctx) {
    __shared__ __bf16 lP[4][16][72];              // per-wave P tile, padded stride
    const int tid = threadIdx.x, lane = tid & 31, wv = tid >> 5;
    const int qb = blockIdx.x, hh = blockIdx.y, b = blockIdx.z;
    const int bh = b * H + hh;
    const int col = lane & 15, halfOff = (lane < 16) ? 0 : 8;

    const __bf16* qbase = q + ((size_t)bh * NCTX + qb * 64 + wv * 16) * DH;
    const v16bf qf0 = frag_MK(qbase, DH, lane);
    const v16bf qf1 = frag_MK(qbase + 32, DH, lane);
    const __bf16* kbase = k + (size_t)bh * NCTX * DH;
    const __bf16* vbase = vt + (size_t)bh * DH * NCTX;

    v8f cacc[4] = {};
    float rowM[8], rowL[8];
    for (int r = 0; r < 8; ++r) { rowM[r] = -1e30f; rowL[r] = 0.f; }

    for (int kb = 0; kb <= qb; ++kb) {
        if (kb < qb)
            __builtin_prefetch(kbase + (size_t)(kb + 1) * 64 * DH, 0, 1);
        // ---- S = Q K^T for a 64-key block (K frags straight from global) ----
        v8f s[4] = {};
        for (int jn = 0; jn < 4; ++jn) {
            const __bf16* kt = kbase + (size_t)(kb * 64 + jn * 16) * DH;
            s[jn] = wmma_bf16(qf0, frag_NK(kt, DH, lane), s[jn]);
            s[jn] = wmma_bf16(qf1, frag_NK(kt + 32, DH, lane), s[jn]);
        }
        // ---- scale, causal mask, block row-max ----
        const bool diag = (kb == qb);
        float bm[8];
        for (int r = 0; r < 8; ++r) bm[r] = -1e30f;
        for (int jn = 0; jn < 4; ++jn) {
            int kg = jn * 16 + col;
            for (int r = 0; r < 8; ++r) {
                float sv = s[jn][r] * 0.125f;               // 1/sqrt(64)
                if (diag && kg > wv * 16 + r + halfOff) sv = -1e30f;
                s[jn][r] = sv;
                bm[r] = fmaxf(bm[r], sv);
            }
        }
        for (int m = 1; m < 16; m <<= 1)
            for (int r = 0; r < 8; ++r)
                bm[r] = fmaxf(bm[r], __shfl_xor(bm[r], m, 32));
        // ---- online softmax update ----
        float alpha[8], bs[8];
        for (int r = 0; r < 8; ++r) {
            float nm = fmaxf(rowM[r], bm[r]);
            alpha[r] = __expf(rowM[r] - nm);
            rowM[r] = nm;
            bs[r] = 0.f;
        }
        for (int jn = 0; jn < 4; ++jn)
            for (int r = 0; r < 8; ++r) {
                float p = __expf(s[jn][r] - rowM[r]);
                s[jn][r] = p;
                bs[r] += p;
            }
        for (int m = 1; m < 16; m <<= 1)
            for (int r = 0; r < 8; ++r)
                bs[r] += __shfl_xor(bs[r], m, 32);
        for (int r = 0; r < 8; ++r) rowL[r] = rowL[r] * alpha[r] + bs[r];
        for (int jd = 0; jd < 4; ++jd)
            for (int r = 0; r < 8; ++r)
                cacc[jd][r] *= alpha[r];
        // ---- P: C-layout regs -> A-layout via per-wave LDS (same-wave order) --
        for (int jn = 0; jn < 4; ++jn)
            for (int r = 0; r < 8; ++r)
                lP[wv][r + halfOff][jn * 16 + col] = (__bf16)s[jn][r];
        for (int ks = 0; ks < 2; ++ks) {
            v16bf pf = frag_MK(&lP[wv][0][ks * 32], 72, lane);
            for (int jd = 0; jd < 4; ++jd) {
                const __bf16* vp = vbase + (size_t)(jd * 16) * NCTX + kb * 64 + ks * 32;
                cacc[jd] = wmma_bf16(pf, frag_NK(vp, NCTX, lane), cacc[jd]);
            }
        }
    }
    // ---- normalize and store ctx bf16 in [b][n][D] (head-interleaved) -------
    for (int jd = 0; jd < 4; ++jd)
        for (int r = 0; r < 8; ++r) {
            int gmrow = qb * 64 + wv * 16 + r + halfOff;
            float v = cacc[jd][r] / rowL[r];
            size_t idx = (size_t)(b * NCTX + gmrow) * D + hh * DH + jd * 16 + col;
            ctx[idx] = (__bf16)v;
        }
}

// ---------------- output projection: out = ctx @ Wo.T + bo (f32 out) ---------
__global__ void __launch_bounds__(128)
oproj_kernel(const __bf16* __restrict__ ctx, const __bf16* __restrict__ W,
             const float* __restrict__ bias, float* __restrict__ out) {
    __shared__ __bf16 lB[64][56];
    const int tid = threadIdx.x, lane = tid & 31, wv = tid >> 5;
    const int m0 = blockIdx.x * 64, n0 = blockIdx.y * 64;
    v8f acc[4] = {};
    const int lr = tid >> 1, lc = (tid & 1) * 16;
    const __bf16* abase = ctx + (size_t)(m0 + wv * 16) * D;
    const __bf16* wrow  = W + (size_t)(n0 + lr) * D + lc;
    __bf16* ldst = &lB[lr][lc];

    for (int k0 = 0; k0 < D; k0 += 32) {
        __syncthreads();
        async_copy_b128(wrow + k0,     ldst);
        async_copy_b128(wrow + k0 + 8, ldst + 8);
        async_wait0();
        __syncthreads();
        v16bf af = frag_MK(abase + k0, D, lane);
        for (int jn = 0; jn < 4; ++jn)
            acc[jn] = wmma_bf16(af, frag_NK(&lB[jn * 16][0], 56, lane), acc[jn]);
    }

    const int col = lane & 15, halfOff = (lane < 16) ? 0 : 8;
    for (int jn = 0; jn < 4; ++jn)
        for (int r = 0; r < 8; ++r) {
            int gm = m0 + wv * 16 + r + halfOff;
            int gn = n0 + jn * 16 + col;
            out[(size_t)gm * D + gn] = acc[jn][r] + bias[gn];
        }
}

extern "C" void kernel_launch(void* const* d_in, const int* in_sizes, int n_in,
                              void* d_out, int out_size, void* d_ws, size_t ws_size,
                              hipStream_t stream) {
    const float* x  = (const float*)d_in[0];
    const float* Wq = (const float*)d_in[1];
    const float* bq = (const float*)d_in[2];
    const float* Wk = (const float*)d_in[3];
    const float* bk = (const float*)d_in[4];
    const float* Wv = (const float*)d_in[5];
    const float* bv = (const float*)d_in[6];
    const float* Wo = (const float*)d_in[7];
    const float* bo = (const float*)d_in[8];
    float* out = (float*)d_out;

    const size_t nX = (size_t)NB * NCTX * D;               // 4M
    const size_t nW = (size_t)D * D;                       // 1M
    const size_t perTensor = (size_t)NB * H * NCTX * DH;   // 4M

    __bf16* xb  = (__bf16*)d_ws;
    __bf16* wqb = xb  + nX;
    __bf16* wkb = wqb + nW;
    __bf16* wvb = wkb + nW;
    __bf16* wob = wvb + nW;
    __bf16* qw  = wob + nW;
    __bf16* kw  = qw  + perTensor;
    __bf16* vtw = kw  + perTensor;
    __bf16* cw  = vtw + perTensor;                         // total 24M bf16 = 48 MB

    cvt_kernel<<<(int)(nX / 8 / 256), 256, 0, stream>>>(x,  xb,  (int)nX);
    cvt_kernel<<<(int)(nW / 8 / 256), 256, 0, stream>>>(Wq, wqb, (int)nW);
    cvt_kernel<<<(int)(nW / 8 / 256), 256, 0, stream>>>(Wk, wkb, (int)nW);
    cvt_kernel<<<(int)(nW / 8 / 256), 256, 0, stream>>>(Wv, wvb, (int)nW);
    cvt_kernel<<<(int)(nW / 8 / 256), 256, 0, stream>>>(Wo, wob, (int)nW);

    dim3 pg(NB * NCTX / 64, D / 64);                       // (64, 16)
    proj_kernel<<<pg, 128, 0, stream>>>(xb, wqb, bq, qw, 0);
    proj_kernel<<<pg, 128, 0, stream>>>(xb, wkb, bk, kw, 0);
    proj_kernel<<<pg, 128, 0, stream>>>(xb, wvb, bv, vtw, 1);
    attn_kernel<<<dim3(NCTX / 64, H, NB), 128, 0, stream>>>(qw, kw, vtw, cw);
    oproj_kernel<<<pg, 128, 0, stream>>>(cw, wob, bo, out);
}
